// EfficientMemoryGEMMWithSoftmax_70781061038962
// MI455X (gfx1250) — matrix operand order repaired and checked by reading, not compiled, not run on hardware
//
#include <hip/hip_runtime.h>
#include <cstddef>
#include <cstdint>

// ---------------------------------------------------------------------------
// Types for CDNA5 WMMA + TDM descriptor groups
// ---------------------------------------------------------------------------
typedef __attribute__((ext_vector_type(16))) __bf16 v16bf;
typedef __attribute__((ext_vector_type(8)))  __bf16 v8bf;
typedef __attribute__((ext_vector_type(8)))  float  v8f;

typedef unsigned tdm_u4 __attribute__((vector_size(16)));
typedef int      tdm_i4 __attribute__((vector_size(16)));
typedef int      tdm_i8 __attribute__((vector_size(32)));

#define AS3 __attribute__((address_space(3)))

// ---------------------------------------------------------------------------
// Problem constants: x1 (2,16,1024,1024) f32, x2 (2,16,1024,64) f32
// ---------------------------------------------------------------------------
#define NBATCH 32
#define MDIM   1024
#define KDIM   1024
#define NDIM   64
#define N_X1   (2ll*16*1024*1024)
#define N_X10  (16ll*1024*1024)
#define N_X2   (2ll*16*1024*64)
#define MROWS  2048
#define MCOLS  1024
#define RANK   16

#define OUT_RES 0ll
#define OUT_X1H (32ll*1024*64)
#define OUT_X2H (OUT_X1H + N_X1)

#define NBINS   8192
#define WS_H1   16
#define WS_H2   (WS_H1 + NBINS)
#define WS_V    (WS_H2 + NBINS)
#define WS_Y    (WS_V + MCOLS*RANK)
#define WS_T    (WS_Y + MROWS*RANK)

// ---------------------------------------------------------------------------
// Helpers
// ---------------------------------------------------------------------------
// split each f32 of a float4 into hi/lo bf16 planes:
//   hi = truncate-to-bf16(f), rem = f - hi (exact), lo = truncate-to-bf16(rem)
__device__ __forceinline__ void split4(const float4& v, ushort4& h, ushort4& l) {
  unsigned u0 = __builtin_bit_cast(unsigned, v.x);
  unsigned u1 = __builtin_bit_cast(unsigned, v.y);
  unsigned u2 = __builtin_bit_cast(unsigned, v.z);
  unsigned u3 = __builtin_bit_cast(unsigned, v.w);
  float r0 = v.x - __builtin_bit_cast(float, u0 & 0xFFFF0000u);
  float r1 = v.y - __builtin_bit_cast(float, u1 & 0xFFFF0000u);
  float r2 = v.z - __builtin_bit_cast(float, u2 & 0xFFFF0000u);
  float r3 = v.w - __builtin_bit_cast(float, u3 & 0xFFFF0000u);
  h = make_ushort4((unsigned short)(u0 >> 16), (unsigned short)(u1 >> 16),
                   (unsigned short)(u2 >> 16), (unsigned short)(u3 >> 16));
  l = make_ushort4((unsigned short)(__builtin_bit_cast(unsigned, r0) >> 16),
                   (unsigned short)(__builtin_bit_cast(unsigned, r1) >> 16),
                   (unsigned short)(__builtin_bit_cast(unsigned, r2) >> 16),
                   (unsigned short)(__builtin_bit_cast(unsigned, r3) >> 16));
}

__device__ __forceinline__ v16bf ld_frag16(const unsigned short* p0,
                                           const unsigned short* p1) {
  v8bf a = *(const v8bf*)p0;
  v8bf b = *(const v8bf*)p1;
  return __builtin_shufflevector(a, b, 0, 1, 2, 3, 4, 5, 6, 7,
                                 8, 9, 10, 11, 12, 13, 14, 15);
}

__device__ __forceinline__ float sub_val(const float* __restrict__ x2, float thr2,
                                         int r, int c) {
  int b = r >> 10, s = r & 1023, h = c >> 6, d = c & 63;
  float v = x2[((((size_t)b * 16 + h) * 1024 + s) * 64) + d];
  return (fabsf(v) > thr2) ? 0.0f : v;
}
__device__ __forceinline__ size_t x2_flat(int r, int c) {
  int b = r >> 10, s = r & 1023, h = c >> 6, d = c & 63;
  return ((((size_t)b * 16 + h) * 1024 + s) * 64) + d;
}

// ---------------------------------------------------------------------------
// TDM demo/prefetch: DMA one 32x64 f32 panel of x2 into LDS via the Tensor
// Data Mover, synchronized with TENSORcnt. D# built per ISA 08_async_tensor:
//   group0: [1:0]=count=1, [63:32]=lds_addr, [120:64]=global_addr, [127:126]=2
//   group1: [17:16]=data_size(2 -> 4B), tensor_dim0=64, tensor_dim1=1024,
//           tile_dim0=64, tile_dim1=32, tensor_dim0_stride=64
//   groups 2/3 + trailing i32x8: zero (2D tensor), cpol=0
// All operands are workgroup-uniform -> SGPR groups, issued by wave 0 only.
// ---------------------------------------------------------------------------
__global__ __launch_bounds__(128) void k_tdm_prefetch(const float* __restrict__ B,
                                                      float* __restrict__ ws) {
  __shared__ __align__(16) float panel[32 * 64];
  const int batch = blockIdx.x;
  unsigned long long gaddr =
      (unsigned long long)(const void*)(B + (size_t)batch * KDIM * NDIM);
  unsigned lds_off = (unsigned)(size_t)((AS3 void*)panel);

  tdm_u4 g0 = { 1u,                                        // count=1, flags 0
                lds_off,                                   // lds_addr (bytes)
                (unsigned)(gaddr & 0xFFFFFFFFull),         // global_addr[31:0]
                (unsigned)((gaddr >> 32) & 0x01FFFFFFull)  // global_addr[56:32]
                    | (2u << 30) };                        // type=2 (image)
  tdm_i8 g1 = { (int)(2u << 16),       // data_size=2 (4 bytes), mask/flags 0
                (int)(64u << 16),      // tensor_dim0[15:0] in bits[63:48]
                (int)(1024u << 16),    // tensor_dim1[15:0] in bits[95:80]
                (int)(64u << 16),      // tile_dim0 in bits[127:112]
                32,                    // tile_dim1 in bits[143:128]
                64,                    // tensor_dim0_stride[31:0]
                0, 0 };
  tdm_i4 g2 = { 0, 0, 0, 0 };
  tdm_i4 g3 = { 0, 0, 0, 0 };
  tdm_i8 g4 = { 0, 0, 0, 0, 0, 0, 0, 0 };

  if (threadIdx.x < 32) {                       // one TDM op per workgroup
    __builtin_amdgcn_tensor_load_to_lds(g0, g1, g2, g3, g4, 0);
    __builtin_amdgcn_s_wait_tensorcnt(0);       // DMA complete -> LDS valid
  }
  __syncthreads();
  if (threadIdx.x == 0) ws[8 + (batch & 3)] = panel[0];   // witness store
}

// ---------------------------------------------------------------------------
// Kernel 1: batched GEMM  result = x1 @ x2  via bf16x3 split-precision WMMA.
//   Software-pipelined: global loads for tile i+1 are issued before the WMMAs
//   of tile i; LDS bf16 planes are double-buffered, one barrier per K-step.
// ---------------------------------------------------------------------------
#define KT 32
#define MT 64
#define STRH 40   // halves per LDS row (80 B, 16B-aligned, 20-dword stagger)
#define NIT (KDIM / KT)

__global__ __launch_bounds__(128) void k_gemm_wmma(const float* __restrict__ A,
                                                   const float* __restrict__ B,
                                                   float* __restrict__ C) {
  __shared__ __align__(16) unsigned short Ah[2][MT * STRH];
  __shared__ __align__(16) unsigned short Al[2][MT * STRH];
  __shared__ __align__(16) unsigned short Bth[2][NDIM * STRH];
  __shared__ __align__(16) unsigned short Btl[2][NDIM * STRH];

  const int tid   = threadIdx.x;
  const int lane  = tid & 31;
  const int wave  = tid >> 5;
  const int mblk  = blockIdx.x;     // 0..15
  const int batch = blockIdx.y;     // 0..31

  const float* Ab = A + (size_t)batch * MDIM * KDIM + (size_t)mblk * MT * KDIM;
  const float* Bb = B + (size_t)batch * KDIM * NDIM;
  float*       Cb = C + (size_t)batch * MDIM * NDIM + (size_t)mblk * MT * NDIM;

  v8f acc[4] = {};

  const int laneHi = lane >> 4;
  const int mrow   = wave * 16 + (lane & 15);  // 16-bit A 16x32 VGPR map
  const int klo    = laneHi ? 8 : 0;
  const int kbofB  = laneHi ? 16 : 0;          // B: lane half h takes k = 16h..16h+15 (same contiguous order as A)
  const int ncol   = lane & 15;

  float4 ra[4], rb[4];   // in-flight K tile

  auto load_regs = [&](int kb) {
#pragma unroll
    for (int i = 0; i < 4; ++i) {
      int idx = tid + i * 128;             // 0..511
      int row = idx >> 3;                  // A: 8 float4 per row
      int c4  = (idx & 7) * 4;
      ra[i] = *(const float4*)(Ab + (size_t)row * KDIM + kb + c4);
    }
#pragma unroll
    for (int i = 0; i < 4; ++i) {
      int idx = tid + i * 128;
      int n  = idx & 63;                   // coalesced over n
      int k0 = (idx >> 6) * 4;
      rb[i] = make_float4(Bb[(size_t)(kb + k0 + 0) * NDIM + n],
                          Bb[(size_t)(kb + k0 + 1) * NDIM + n],
                          Bb[(size_t)(kb + k0 + 2) * NDIM + n],
                          Bb[(size_t)(kb + k0 + 3) * NDIM + n]);
    }
  };

  auto store_lds = [&](int pb) {
#pragma unroll
    for (int i = 0; i < 4; ++i) {
      int idx = tid + i * 128;
      int row = idx >> 3;
      int c4  = (idx & 7) * 4;
      ushort4 h, l; split4(ra[i], h, l);
      *(ushort4*)(&Ah[pb][row * STRH + c4]) = h;
      *(ushort4*)(&Al[pb][row * STRH + c4]) = l;
    }
#pragma unroll
    for (int i = 0; i < 4; ++i) {
      int idx = tid + i * 128;
      int n  = idx & 63;
      int k0 = (idx >> 6) * 4;
      ushort4 h, l; split4(rb[i], h, l);
      *(ushort4*)(&Bth[pb][n * STRH + k0]) = h;
      *(ushort4*)(&Btl[pb][n * STRH + k0]) = l;
    }
  };

  load_regs(0);
  store_lds(0);

  for (int it = 0; it < NIT; ++it) {
    __syncthreads();
    const int pb = it & 1;
    if (it + 1 < NIT) load_regs((it + 1) * KT);   // issue next tile's loads

    const unsigned short* arh = &Ah[pb][mrow * STRH];
    const unsigned short* arl = &Al[pb][mrow * STRH];
    v16bf a_hi = ld_frag16(arh + 2 * klo, arh + 2 * klo + 8);
    v16bf a_lo = ld_frag16(arl + 2 * klo, arl + 2 * klo + 8);
#pragma unroll
    for (int nt = 0; nt < 4; ++nt) {
      const unsigned short* brh = &Bth[pb][(nt * 16 + ncol) * STRH + kbofB];
      const unsigned short* brl = &Btl[pb][(nt * 16 + ncol) * STRH + kbofB];
      v16bf b_hi = ld_frag16(brh, brh + 8);
      v16bf b_lo = ld_frag16(brl, brl + 8);
      acc[nt] = __builtin_amdgcn_wmma_f32_16x16x32_bf16(false, a_hi, false, b_hi,
                                                        (short)0, acc[nt], false, false);
      acc[nt] = __builtin_amdgcn_wmma_f32_16x16x32_bf16(false, a_hi, false, b_lo,
                                                        (short)0, acc[nt], false, false);
      acc[nt] = __builtin_amdgcn_wmma_f32_16x16x32_bf16(false, a_lo, false, b_hi,
                                                        (short)0, acc[nt], false, false);
    }
    if (it + 1 < NIT) store_lds(pb ^ 1);          // park next tile
  }

  const int rhi = laneHi ? 8 : 0;
#pragma unroll
  for (int nt = 0; nt < 4; ++nt)
#pragma unroll
    for (int j = 0; j < 8; ++j)
      Cb[(size_t)(wave * 16 + rhi + j) * NDIM + nt * 16 + ncol] = acc[nt][j];
}

// ---------------------------------------------------------------------------
// Workspace init
// ---------------------------------------------------------------------------
__global__ void k_init_ws(unsigned* __restrict__ U) {
  int i = blockIdx.x * blockDim.x + threadIdx.x;
  if (i < 2 * NBINS) U[WS_H1 + i] = 0u;
  if (i == 0) {
    U[0] = 0x7F7FFFFFu;
    U[1] = 0u;
    U[2] = 0u;
    U[5] = 0u;
  }
}

// ---------------------------------------------------------------------------
// x1 quantile path
// ---------------------------------------------------------------------------
__global__ void k_minmax_x1(const float* __restrict__ x1, unsigned* __restrict__ U) {
  size_t i = (size_t)blockIdx.x * blockDim.x + threadIdx.x;
  size_t stride = (size_t)gridDim.x * blockDim.x;
  float mn = 3.4e38f, mx = 0.0f;
  for (; i < N_X10; i += stride) {
    float v = x1[i];
    mn = fminf(mn, v); mx = fmaxf(mx, v);
  }
  atomicMin(&U[0], __builtin_bit_cast(unsigned, mn));
  atomicMax(&U[1], __builtin_bit_cast(unsigned, mx));
}

__global__ __launch_bounds__(256) void k_hist1(const float* __restrict__ x1,
                                               unsigned* __restrict__ U) {
  __shared__ unsigned hl[NBINS];
  float lo = __builtin_bit_cast(float, U[0]);
  float hi = __builtin_bit_cast(float, U[1]);
  float invw = (hi > lo) ? (float)NBINS / (hi - lo) : 0.0f;
  for (int i = threadIdx.x; i < NBINS; i += 256) hl[i] = 0u;
  __syncthreads();
  size_t i = (size_t)blockIdx.x * blockDim.x + threadIdx.x;
  size_t stride = (size_t)gridDim.x * blockDim.x;
  for (; i < N_X10; i += stride) {
    int b = (int)((x1[i] - lo) * invw);
    b = (b < 0) ? 0 : ((b > NBINS - 1) ? NBINS - 1 : b);
    atomicAdd(&hl[b], 1u);
  }
  __syncthreads();
  for (int i = threadIdx.x; i < NBINS; i += 256)
    if (hl[i]) atomicAdd(&U[WS_H1 + i], hl[i]);
}

__global__ void k_scan1(float* __restrict__ ws) {
  unsigned* U = (unsigned*)ws;
  float lo = __builtin_bit_cast(float, U[0]);
  float hi = __builtin_bit_cast(float, U[1]);
  long long rank = (long long)(0.99 * (double)N_X10);
  long long cum = 0; int bin = NBINS - 1;
  for (int i = 0; i < NBINS; ++i) { cum += U[WS_H1 + i]; if (cum >= rank) { bin = i; break; } }
  ws[3] = lo + (hi - lo) * (float)(bin + 1) / (float)NBINS;
}

__global__ void k_x1hat(const float* __restrict__ x1, const float* __restrict__ ws,
                        float* __restrict__ out2) {
  float thr = ws[3];
  size_t i = (size_t)blockIdx.x * blockDim.x + threadIdx.x;
  size_t stride = (size_t)gridDim.x * blockDim.x;
  const float4* in4 = (const float4*)x1;
  float4* o4 = (float4*)out2;
  for (; i < (size_t)(N_X1 / 4); i += stride) {
    float4 v = in4[i];
    v.x = (v.x > thr) ? v.x : 0.0f;
    v.y = (v.y > thr) ? v.y : 0.0f;
    v.z = (v.z > thr) ? v.z : 0.0f;
    v.w = (v.w > thr) ? v.w : 0.0f;
    o4[i] = v;
  }
}

// ---------------------------------------------------------------------------
// x2 quantile path
// ---------------------------------------------------------------------------
__global__ void k_absmax_x2(const float* __restrict__ x2, unsigned* __restrict__ U) {
  size_t i = (size_t)blockIdx.x * blockDim.x + threadIdx.x;
  size_t stride = (size_t)gridDim.x * blockDim.x;
  float mx = 0.0f;
  for (; i < N_X2; i += stride) mx = fmaxf(mx, fabsf(x2[i]));
  atomicMax(&U[2], __builtin_bit_cast(unsigned, mx));
}

__global__ __launch_bounds__(256) void k_hist2(const float* __restrict__ x2,
                                               unsigned* __restrict__ U) {
  __shared__ unsigned hl[NBINS];
  float hi = __builtin_bit_cast(float, U[2]);
  float invw = (hi > 0.0f) ? (float)NBINS / hi : 0.0f;
  for (int i = threadIdx.x; i < NBINS; i += 256) hl[i] = 0u;
  __syncthreads();
  size_t i = (size_t)blockIdx.x * blockDim.x + threadIdx.x;
  size_t stride = (size_t)gridDim.x * blockDim.x;
  for (; i < N_X2; i += stride) {
    int b = (int)(fabsf(x2[i]) * invw);
    b = (b < 0) ? 0 : ((b > NBINS - 1) ? NBINS - 1 : b);
    atomicAdd(&hl[b], 1u);
  }
  __syncthreads();
  for (int i = threadIdx.x; i < NBINS; i += 256)
    if (hl[i]) atomicAdd(&U[WS_H2 + i], hl[i]);
}

__global__ void k_scan2(float* __restrict__ ws) {
  unsigned* U = (unsigned*)ws;
  float hi = __builtin_bit_cast(float, U[2]);
  long long rank = (long long)(0.99 * (double)N_X2);
  long long cum = 0; int bin = NBINS - 1;
  for (int i = 0; i < NBINS; ++i) { cum += U[WS_H2 + i]; if (cum >= rank) { bin = i; break; } }
  ws[4] = hi * (float)(bin + 1) / (float)NBINS;
}

// ---------------------------------------------------------------------------
// rank-16 subspace iteration on sub = m - outliers  (2048 x 1024)
// ---------------------------------------------------------------------------
__global__ void k_vinit(float* __restrict__ ws) {
  int i = blockIdx.x * blockDim.x + threadIdx.x;
  if (i < MCOLS * RANK) {
    unsigned u = (unsigned)i * 2654435761u;
    u ^= u >> 16; u *= 2246822519u; u ^= u >> 13;
    ws[WS_V + i] = (float)(u & 0xFFFFFFu) * (2.0f / 16777216.0f) - 1.0f;
  }
}

__global__ __launch_bounds__(256) void k_subV(const float* __restrict__ x2,
                                              float* __restrict__ ws) {
  __shared__ float accv[RANK];
  const int tid = threadIdx.x;
  const int r = blockIdx.x;
  const float thr2 = ws[4];
  const float* V = ws + WS_V;
  float* Y = ws + WS_Y;
  if (tid < RANK) accv[tid] = 0.0f;
  __syncthreads();
  float loc[RANK];
#pragma unroll
  for (int j = 0; j < RANK; ++j) loc[j] = 0.0f;
  for (int c = tid; c < MCOLS; c += 256) {
    float s = sub_val(x2, thr2, r, c);
#pragma unroll
    for (int j = 0; j < RANK; ++j) loc[j] += s * V[c * RANK + j];
  }
#pragma unroll
  for (int j = 0; j < RANK; ++j) atomicAdd(&accv[j], loc[j]);   // ds_add_f32
  __syncthreads();
  if (tid < RANK) Y[r * RANK + tid] = accv[tid];
}

__global__ __launch_bounds__(256) void k_subTY(const float* __restrict__ x2,
                                               float* __restrict__ ws) {
  __shared__ float accv[RANK];
  const int tid = threadIdx.x;
  const int c = blockIdx.x;
  const float thr2 = ws[4];
  const float* Y = ws + WS_Y;
  float* T = ws + WS_T;
  if (tid < RANK) accv[tid] = 0.0f;
  __syncthreads();
  float loc[RANK];
#pragma unroll
  for (int j = 0; j < RANK; ++j) loc[j] = 0.0f;
  for (int r = tid; r < MROWS; r += 256) {
    float s = sub_val(x2, thr2, r, c);
#pragma unroll
    for (int j = 0; j < RANK; ++j) loc[j] += s * Y[r * RANK + j];
  }
#pragma unroll
  for (int j = 0; j < RANK; ++j) atomicAdd(&accv[j], loc[j]);
  __syncthreads();
  if (tid < RANK) T[c * RANK + tid] = accv[tid];
}

__global__ __launch_bounds__(256) void k_orth(float* __restrict__ ws) {
  __shared__ float red[256];
  const int tid = threadIdx.x;
  float* V = ws + WS_V;
  float* T = ws + WS_T;
  auto blockReduce = [&](float v) -> float {
    red[tid] = v; __syncthreads();
    for (int s = 128; s > 0; s >>= 1) {
      if (tid < s) red[tid] += red[tid + s];
      __syncthreads();
    }
    float r = red[0]; __syncthreads();
    return r;
  };
  for (int j = 0; j < RANK; ++j) {
    for (int k = 0; k < j; ++k) {
      float local = 0.0f;
      for (int c = tid; c < MCOLS; c += 256) local += T[c * RANK + j] * V[c * RANK + k];
      float d = blockReduce(local);
      for (int c = tid; c < MCOLS; c += 256) T[c * RANK + j] -= d * V[c * RANK + k];
      __syncthreads();
    }
    float local = 0.0f;
    for (int c = tid; c < MCOLS; c += 256) { float t = T[c * RANK + j]; local += t * t; }
    float nrm = sqrtf(blockReduce(local)) + 1e-20f;
    for (int c = tid; c < MCOLS; c += 256) V[c * RANK + j] = T[c * RANK + j] / nrm;
    __syncthreads();
  }
}

__global__ void k_residmax(const float* __restrict__ x2, float* __restrict__ ws) {
  unsigned* U = (unsigned*)ws;
  const float thr2 = ws[4];
  const float* V = ws + WS_V;
  const float* Y = ws + WS_Y;
  size_t i = (size_t)blockIdx.x * blockDim.x + threadIdx.x;
  size_t stride = (size_t)gridDim.x * blockDim.x;
  float mx = 0.0f;
  for (; i < (size_t)MROWS * MCOLS; i += stride) {
    int r = (int)(i >> 10), c = (int)(i & 1023);
    float s = sub_val(x2, thr2, r, c);
    float lr = 0.0f;
#pragma unroll
    for (int j = 0; j < RANK; ++j) lr += Y[r * RANK + j] * V[c * RANK + j];
    mx = fmaxf(mx, fabsf(s - lr));
  }
  atomicMax(&U[5], __builtin_bit_cast(unsigned, mx));
}

__global__ void k_scale(float* __restrict__ ws) {
  unsigned* U = (unsigned*)ws;
  ws[6] = __builtin_bit_cast(float, U[5]) / 127.0f + 1e-12f;
}

__global__ void k_rec(const float* __restrict__ x2, const float* __restrict__ ws,
                      float* __restrict__ out3) {
  const float thr2 = ws[4];
  const float scale = ws[6];
  const float* V = ws + WS_V;
  const float* Y = ws + WS_Y;
  size_t i = (size_t)blockIdx.x * blockDim.x + threadIdx.x;
  size_t stride = (size_t)gridDim.x * blockDim.x;
  for (; i < (size_t)MROWS * MCOLS; i += stride) {
    int r = (int)(i >> 10), c = (int)(i & 1023);
    size_t idx = x2_flat(r, c);
    float v = x2[idx];
    float outl = (fabsf(v) > thr2) ? v : 0.0f;
    float s = v - outl;
    float lr = 0.0f;
#pragma unroll
    for (int j = 0; j < RANK; ++j) lr += Y[r * RANK + j] * V[c * RANK + j];
    float resid = s - lr;
    float q = fminf(fmaxf(rintf(resid / scale), -128.0f), 127.0f);
    out3[idx] = outl + q * scale + lr;
  }
}

// ---------------------------------------------------------------------------
// Launch
// ---------------------------------------------------------------------------
extern "C" void kernel_launch(void* const* d_in, const int* in_sizes, int n_in,
                              void* d_out, int out_size, void* d_ws, size_t ws_size,
                              hipStream_t stream) {
  (void)in_sizes; (void)n_in; (void)out_size; (void)ws_size;
  const float* x1 = (const float*)d_in[0];
  const float* x2 = (const float*)d_in[1];
  float* out = (float*)d_out;
  float* ws  = (float*)d_ws;
  unsigned* wsu = (unsigned*)d_ws;

  k_init_ws<<<(2 * NBINS + 255) / 256, 256, 0, stream>>>(wsu);

  // TDM warm-up: stream each batch's leading B panel through the Tensor DMA
  k_tdm_prefetch<<<NBATCH, 128, 0, stream>>>(x2, ws);

  dim3 ggrid(MDIM / MT, NBATCH);
  k_gemm_wmma<<<ggrid, 128, 0, stream>>>(x1, x2, out + OUT_RES);

  k_minmax_x1<<<2048, 256, 0, stream>>>(x1, wsu);
  k_hist1<<<2048, 256, 0, stream>>>(x1, wsu);
  k_scan1<<<1, 1, 0, stream>>>(ws);
  k_x1hat<<<4096, 256, 0, stream>>>(x1, ws, out + OUT_X1H);

  k_absmax_x2<<<1024, 256, 0, stream>>>(x2, wsu);
  k_hist2<<<1024, 256, 0, stream>>>(x2, wsu);
  k_scan2<<<1, 1, 0, stream>>>(ws);
  k_vinit<<<(MCOLS * RANK + 255) / 256, 256, 0, stream>>>(ws);
  for (int it = 0; it < 6; ++it) {
    k_subV <<<MROWS, 256, 0, stream>>>(x2, ws);
    k_subTY<<<MCOLS, 256, 0, stream>>>(x2, ws);
    k_orth <<<1,     256, 0, stream>>>(ws);
  }
  k_subV<<<MROWS, 256, 0, stream>>>(x2, ws);
  k_residmax<<<1024, 256, 0, stream>>>(x2, ws);
  k_scale<<<1, 1, 0, stream>>>(ws);
  k_rec<<<2048, 256, 0, stream>>>(x2, ws, out + OUT_X2H);
}
